// ThinStackHybridLSTM_44427141710410
// MI455X (gfx1250) — matrix operand order repaired and checked by reading, not compiled, not run on hardware
//
#include <hip/hip_runtime.h>
#include <hip/hip_bf16.h>

typedef __bf16 bf16_t;
typedef __attribute__((ext_vector_type(16))) bf16_t v16bf;
typedef __attribute__((ext_vector_type(8)))  bf16_t v8bf;
typedef __attribute__((ext_vector_type(8)))  float  v8f;

#define Bq   1024
#define Lq   64
#define NTRq 127            // 2*L-1
#define Eq   256
#define Hq   256
#define Tq   256
#define Oq   3
#define SSq  ((Lq + 2) * Hq)   // stack row stride (entries * H)
#define GN   (4 * Tq)          // 1024 gate width
#define RN   (5 * Hq)          // 1280 reduce width

// LDS B staging: 64 rows x 32 halves per k-step, padded to 40 halves/row
// (80B row stride -> consuming lanes land on distinct banks). Two buffers.
#define BROW 40
#define BBUF (64 * BROW)       // halves per buffer

#if __has_builtin(__builtin_amdgcn_global_load_async_to_lds_b128)
#define ASYNC_STAGE 1
typedef int v4i_t __attribute__((ext_vector_type(4)));
typedef __attribute__((address_space(1))) v4i_t* gptr_v4i;
typedef __attribute__((address_space(3))) v4i_t* lptr_v4i;
#endif

#define WMMA_BF16(a, b, c) \
  __builtin_amdgcn_wmma_f32_16x16x32_bf16(false, (a), false, (b), (short)0, (c), false, false)

// ---------------------------------------------------------------- helpers
__device__ __forceinline__ float sigm(float x) { return 1.0f / (1.0f + expf(-x)); }
__device__ __forceinline__ float selu_f(float x) {
  const float l = 1.0507009873554805f, a = 1.6732632423543772f;
  return x > 0.0f ? l * x : l * a * (expf(x) - 1.0f);
}

// A-operand tile 16x32 bf16 (M x K): lanes 0-15 -> M rows; lane-half selects
// koff 0/8, second K block at +16 (per ISA A-matrix striping).
__device__ __forceinline__ v16bf load_tile_A(const bf16_t* __restrict__ rowbase,
                                             int lda, int k0, int lane) {
  int r    = lane & 15;
  int koff = (lane >> 4) << 3;           // 0 or 8
  const bf16_t* q = rowbase + r * lda + k0 + koff;
  union { v16bf v; v8bf h[2]; } u;
  u.h[0] = *reinterpret_cast<const v8bf*>(q);
  u.h[1] = *reinterpret_cast<const v8bf*>(q + 16);
  return u.v;
}

// Stage one 64x32 B strip (rows n0..n0+63, k0..k0+31 of row-major N x K W)
// into LDS. 256 threads x one b128 each. Async global->LDS when available.
__device__ __forceinline__ void stage_B(const bf16_t* __restrict__ W, int ldw,
                                        int n0, int k0, bf16_t* buf, int tid) {
  int row = tid >> 2;                 // 0..63
  int ch  = (tid & 3) << 3;           // 0,8,16,24
  const bf16_t* src = W + (n0 + row) * ldw + k0 + ch;
  bf16_t* dst = buf + row * BROW + ch;
#ifdef ASYNC_STAGE
  __builtin_amdgcn_global_load_async_to_lds_b128(
      (gptr_v4i)(void*)src, (lptr_v4i)(void*)dst, 0, 0);
#else
  *reinterpret_cast<v8bf*>(dst) = *reinterpret_cast<const v8bf*>(src);
#endif
}

__device__ __forceinline__ void wait_stage() {
#ifdef ASYNC_STAGE
#if __has_builtin(__builtin_amdgcn_s_wait_asynccnt)
  __builtin_amdgcn_s_wait_asynccnt(0);
#else
  asm volatile("s_wait_asynccnt 0" ::: "memory");
#endif
#endif
  __syncthreads();
}

// B-operand tile 32x16 from LDS strip: lane holds n = lane&15, K at
// 8*(lane>>4)+0..7 and 16+8*(lane>>4)+0..7  (ds_load_b128 x2).
__device__ __forceinline__ v16bf read_B_lds(const bf16_t* buf, int j, int lane) {
  const bf16_t* q = buf + ((j << 4) + (lane & 15)) * BROW + ((lane >> 4) << 3);
  union { v16bf v; v8bf h[2]; } u;
  u.h[0] = *reinterpret_cast<const v8bf*>(q);
  u.h[1] = *reinterpret_cast<const v8bf*>(q + 16);
  return u.v;
}

// One K-segment of C += A @ W^T with LDS-staged, double-buffered B.
__device__ __forceinline__ void gemm_seg_lds(const bf16_t* __restrict__ A, int lda,
                                             const bf16_t* __restrict__ W, int ldw,
                                             int K, int m0, int n0,
                                             int lane, int tid,
                                             bf16_t* sbuf, v8f acc[4]) {
  const bf16_t* rowbase = A + m0 * lda;
  int cur = 0;
  stage_B(W, ldw, n0, 0, sbuf, tid);
  wait_stage();
  for (int k0 = 0; k0 < K; k0 += 32) {
    if (k0 + 32 < K) stage_B(W, ldw, n0, k0 + 32, sbuf + (cur ^ 1) * BBUF, tid);
    v16bf a = load_tile_A(rowbase, lda, k0, lane);
    bf16_t* bb = sbuf + cur * BBUF;
#pragma unroll
    for (int j = 0; j < 4; ++j) {
      v16bf b = read_B_lds(bb, j, lane);
      acc[j] = WMMA_BF16(a, b, acc[j]);
    }
    wait_stage();
    cur ^= 1;
  }
}

// ---------------------------------------------------------------- kernels
__global__ void k_convert_bf16(const float* __restrict__ src,
                               bf16_t* __restrict__ dst, int n) {
  int i = blockIdx.x * blockDim.x + threadIdx.x;
  if (i < n) dst[i] = (bf16_t)src[i];
}

// Phase 1: buf = selu(embed[tokens] @ W_in^T + b_in), stored bf16.
// Block = 128 rows x 64 cols; A rows gathered from f32 embedding table.
__global__ void k_embed_mlp(const int* __restrict__ tokens,
                            const float* __restrict__ embed,
                            const bf16_t* __restrict__ Win,   // H x E bf16
                            const float* __restrict__ bin,    // H
                            bf16_t* __restrict__ bufH) {      // (B*NTR) x H
  __shared__ bf16_t sbuf[2 * BBUF];
  int tid  = threadIdx.x;
  int lane = tid & 31;
  const int nblkN = Hq / 64;                     // 4
  int mb = blockIdx.x / nblkN, nb = blockIdx.x % nblkN;
  int m0 = mb * 128 + (tid >> 5) * 16;
  int n0 = nb * 64;

  v8f acc[4] = {};
  int r    = lane & 15;
  int koff = (lane >> 4) << 3;
  long long tok = tokens[m0 + r];
  const float* arow = embed + tok * Eq;

  int cur = 0;
  stage_B(Win, Eq, n0, 0, sbuf, tid);
  wait_stage();
  for (int k0 = 0; k0 < Eq; k0 += 32) {
    if (k0 + 32 < Eq) stage_B(Win, Eq, n0, k0 + 32, sbuf + (cur ^ 1) * BBUF, tid);
    v16bf a;
    const float* q = arow + k0 + koff;
#pragma unroll
    for (int i = 0; i < 8; ++i) {
      a[i]     = (bf16_t)q[i];
      a[8 + i] = (bf16_t)q[16 + i];
    }
    bf16_t* bb = sbuf + cur * BBUF;
#pragma unroll
    for (int j = 0; j < 4; ++j) {
      v16bf b = read_B_lds(bb, j, lane);
      acc[j] = WMMA_BF16(a, b, acc[j]);
    }
    wait_stage();
    cur ^= 1;
  }

  int hi = lane >> 4, nl = lane & 15;
#pragma unroll
  for (int j = 0; j < 4; ++j) {
    int n = n0 + 16 * j + nl;
    float bb = bin[n];
#pragma unroll
    for (int v = 0; v < 8; ++v) {
      int m = m0 + v + 8 * hi;
      bufH[m * Hq + n] = (bf16_t)selu_f(acc[j][v] + bb);
    }
  }
}

// outputs0 = log_softmax(buf @ W_out^T + b_out), laid out (NTR, B, 3)
__global__ void k_out0(const bf16_t* __restrict__ bufH,
                       const float* __restrict__ Wout,  // 3 x H f32
                       const float* __restrict__ bout,
                       float* __restrict__ out0) {
  int m = blockIdx.x * blockDim.x + threadIdx.x;
  if (m >= Bq * NTRq) return;
  int b = m / NTRq, t = m % NTRq;
  const bf16_t* h = bufH + m * Hq;
  float z[Oq];
#pragma unroll
  for (int o = 0; o < Oq; ++o) {
    float s = bout[o];
    const float* w = Wout + o * Hq;
    for (int k = 0; k < Hq; ++k) s += (float)h[k] * w[k];
    z[o] = s;
  }
  float mx = fmaxf(z[0], fmaxf(z[1], z[2]));
  float lse = logf(expf(z[0] - mx) + expf(z[1] - mx) + expf(z[2] - mx)) + mx;
  float* dst = out0 + ((long long)t * Bq + b) * Oq;
#pragma unroll
  for (int o = 0; o < Oq; ++o) dst[o] = z[o] - lse;
}

// stack[:,0] = stack[:,1] = pad = buf[0, NTR-1] (h==c), th = tc = 0
__global__ void k_init(const bf16_t* __restrict__ bufH,
                       bf16_t* __restrict__ stack_h, float* __restrict__ stack_c,
                       bf16_t* __restrict__ th, float* __restrict__ tc) {
  int i = blockIdx.x * blockDim.x + threadIdx.x;
  if (i >= Bq * Hq) return;
  int b = i / Hq, h = i % Hq;
  bf16_t pad = bufH[(NTRq - 1) * Hq + h];     // batch row 0 only (reference quirk)
  float padf = (float)pad;
  stack_h[b * SSq + h]      = pad;
  stack_h[b * SSq + Hq + h] = pad;
  stack_c[b * SSq + h]      = padf;
  stack_c[b * SSq + Hq + h] = padf;
  th[i] = (bf16_t)0.0f;
  tc[i] = 0.0f;
}

// Generic 4-segment GEMM: C[M x N] = sum_s A_s @ W_s^T + bias0 + bias1
// Block = 128 rows x 64 cols (requires M%128==0, N%64==0).
__global__ void k_gemm4(const bf16_t* A0, int lda0, const bf16_t* W0, int ldw0, int K0,
                        const bf16_t* A1, int lda1, const bf16_t* W1, int ldw1, int K1,
                        const bf16_t* A2, int lda2, const bf16_t* W2, int ldw2, int K2,
                        const bf16_t* A3, int lda3, const bf16_t* W3, int ldw3, int K3,
                        int M, int N,
                        const float* __restrict__ bias0,
                        const float* __restrict__ bias1,
                        float* __restrict__ C) {
  __shared__ bf16_t sbuf[2 * BBUF];
  int tid  = threadIdx.x;
  int lane = tid & 31;
  int nblkN = N >> 6;
  int mb = blockIdx.x / nblkN, nb = blockIdx.x % nblkN;
  int m0 = mb * 128 + (tid >> 5) * 16;
  int n0 = nb * 64;

  v8f acc[4] = {};
  if (K0) gemm_seg_lds(A0, lda0, W0, ldw0, K0, m0, n0, lane, tid, sbuf, acc);
  if (K1) gemm_seg_lds(A1, lda1, W1, ldw1, K1, m0, n0, lane, tid, sbuf, acc);
  if (K2) gemm_seg_lds(A2, lda2, W2, ldw2, K2, m0, n0, lane, tid, sbuf, acc);
  if (K3) gemm_seg_lds(A3, lda3, W3, ldw3, K3, m0, n0, lane, tid, sbuf, acc);

  int hi = lane >> 4, nl = lane & 15;
#pragma unroll
  for (int j = 0; j < 4; ++j) {
    int n = n0 + 16 * j + nl;
    float bb = (bias0 ? bias0[n] : 0.0f) + (bias1 ? bias1[n] : 0.0f);
#pragma unroll
    for (int v = 0; v < 8; ++v) C[(m0 + v + 8 * hi) * N + n] = acc[j][v] + bb;
  }
}

// Tracking LSTM elementwise update
__global__ void k_lstm(const float* __restrict__ gates,
                       float* __restrict__ tc, bf16_t* __restrict__ th) {
  int i = blockIdx.x * blockDim.x + threadIdx.x;
  if (i >= Bq * Tq) return;
  int b = i / Tq, j = i % Tq;
  const float* g = gates + b * GN;
  float i_ = g[j], f_ = g[Tq + j], g_ = g[2 * Tq + j], o_ = g[3 * Tq + j];
  float c = sigm(f_) * tc[i] + sigm(i_) * tanhf(g_);
  tc[i] = c;
  th[i] = (bf16_t)(sigm(o_) * tanhf(c));
}

// SHIFT: stack[:,p] = [buf_k, buf_k]
__global__ void k_shift(const bf16_t* __restrict__ bufH, int k, int p,
                        bf16_t* __restrict__ stack_h, float* __restrict__ stack_c) {
  int i = blockIdx.x * blockDim.x + threadIdx.x;
  if (i >= Bq * Hq) return;
  int b = i / Hq, h = i % Hq;
  bf16_t v = bufH[((long long)b * NTRq + k) * Hq + h];
  stack_h[b * SSq + p * Hq + h] = v;
  stack_c[b * SSq + p * Hq + h] = (float)v;
}

// REDUCE elementwise: tree-LSTM combine, writes stack[p-2]
__global__ void k_reduce(const float* __restrict__ li, int p,
                         bf16_t* __restrict__ stack_h, float* __restrict__ stack_c) {
  int i = blockIdx.x * blockDim.x + threadIdx.x;
  if (i >= Bq * Hq) return;
  int b = i / Hq, h = i % Hq;
  const float* g = li + b * RN;
  float a  = g[h],          ig = g[Hq + h],  f1 = g[2 * Hq + h];
  float f2 = g[3 * Hq + h], og = g[4 * Hq + h];
  float lc = stack_c[b * SSq + (p - 2) * Hq + h];
  float rc = stack_c[b * SSq + (p - 1) * Hq + h];
  float c  = tanhf(a) * sigm(ig) + sigm(f1) * lc + sigm(f2) * rc;
  float hh = sigm(og) * tanhf(c);
  stack_c[b * SSq + (p - 2) * Hq + h] = c;
  stack_h[b * SSq + (p - 2) * Hq + h] = (bf16_t)hh;
}

// outs1[r] = log_softmax(h_new @ W_out^T + b_out)
__global__ void k_out1(const bf16_t* __restrict__ stack_h, int p,
                       const float* __restrict__ Wout, const float* __restrict__ bout,
                       float* __restrict__ out1, int r) {
  int b = blockIdx.x * blockDim.x + threadIdx.x;
  if (b >= Bq) return;
  const bf16_t* h = stack_h + b * SSq + (p - 2) * Hq;
  float z[Oq];
#pragma unroll
  for (int o = 0; o < Oq; ++o) {
    float s = bout[o];
    const float* w = Wout + o * Hq;
    for (int k = 0; k < Hq; ++k) s += (float)h[k] * w[k];
    z[o] = s;
  }
  float mx = fmaxf(z[0], fmaxf(z[1], z[2]));
  float lse = logf(expf(z[0] - mx) + expf(z[1] - mx) + expf(z[2] - mx)) + mx;
  float* dst = out1 + ((long long)r * Bq + b) * Oq;
#pragma unroll
  for (int o = 0; o < Oq; ++o) dst[o] = z[o] - lse;
}

// ---------------------------------------------------------------- host
extern "C" void kernel_launch(void* const* d_in, const int* in_sizes, int n_in,
                              void* d_out, int out_size, void* d_ws, size_t ws_size,
                              hipStream_t stream) {
  const int*   tokens  = (const int*)d_in[0];
  const float* embed   = (const float*)d_in[2];
  const float* W_in_w  = (const float*)d_in[3];
  const float* W_in_b  = (const float*)d_in[4];
  const float* left_w  = (const float*)d_in[5];
  const float* left_b  = (const float*)d_in[6];
  const float* right_w = (const float*)d_in[7];
  const float* track_w = (const float*)d_in[8];
  const float* w_ih    = (const float*)d_in[9];
  const float* w_hh    = (const float*)d_in[10];
  const float* b_ih    = (const float*)d_in[11];
  const float* b_hh    = (const float*)d_in[12];
  const float* W_out_w = (const float*)d_in[13];
  const float* W_out_b = (const float*)d_in[14];

  // workspace layout
  char* wsp = (char*)d_ws;
  auto take = [&](size_t bytes) -> void* {
    void* p = wsp; wsp += (bytes + 255) & ~(size_t)255; return p;
  };
  bf16_t* Win_bf   = (bf16_t*)take((size_t)Hq * Eq * 2);
  bf16_t* Wih_bf   = (bf16_t*)take((size_t)GN * (3 * Hq) * 2);
  bf16_t* Whh_bf   = (bf16_t*)take((size_t)GN * Tq * 2);
  bf16_t* Lw_bf    = (bf16_t*)take((size_t)RN * Hq * 2);
  bf16_t* Rw_bf    = (bf16_t*)take((size_t)RN * Hq * 2);
  bf16_t* Tw_bf    = (bf16_t*)take((size_t)RN * Tq * 2);
  bf16_t* bufH     = (bf16_t*)take((size_t)Bq * NTRq * Hq * 2);
  bf16_t* stack_h  = (bf16_t*)take((size_t)Bq * SSq * 2);
  float*  stack_c  = (float*) take((size_t)Bq * SSq * 4);
  bf16_t* th       = (bf16_t*)take((size_t)Bq * Tq * 2);
  float*  tc       = (float*) take((size_t)Bq * Tq * 4);
  float*  gates    = (float*) take((size_t)Bq * GN * 4);
  float*  lstm_in  = (float*) take((size_t)Bq * RN * 4);
  (void)ws_size; (void)in_sizes; (void)n_in; (void)out_size;

  float* out0 = (float*)d_out;                          // (NTR, B, 3)
  float* out1 = (float*)d_out + (size_t)NTRq * Bq * Oq; // (63, B, 3)

  auto cvt = [&](const float* s, bf16_t* d, int n) {
    k_convert_bf16<<<(n + 255) / 256, 256, 0, stream>>>(s, d, n);
  };
  cvt(W_in_w,  Win_bf, Hq * Eq);
  cvt(w_ih,    Wih_bf, GN * 3 * Hq);
  cvt(w_hh,    Whh_bf, GN * Tq);
  cvt(left_w,  Lw_bf,  RN * Hq);
  cvt(right_w, Rw_bf,  RN * Hq);
  cvt(track_w, Tw_bf,  RN * Tq);

  // Phase 1: buf + outputs0
  {
    int blocks = ((Bq * NTRq) / 128) * (Hq / 64);    // 1016*4 = 4064
    k_embed_mlp<<<blocks, 256, 0, stream>>>(tokens, embed, Win_bf, W_in_b, bufH);
    k_out0<<<(Bq * NTRq + 255) / 256, 256, 0, stream>>>(bufH, W_out_w, W_out_b, out0);
  }
  k_init<<<(Bq * Hq + 255) / 256, 256, 0, stream>>>(bufH, stack_h, stack_c, th, tc);

  // schedule: SHIFT,SHIFT,(SHIFT,REDUCE)*62,REDUCE   (SHIFT=1, REDUCE=2)
  int sched[NTRq];
  { int idx = 0; sched[idx++] = 1; sched[idx++] = 1;
    for (int i = 0; i < Lq - 2; ++i) { sched[idx++] = 1; sched[idx++] = 2; }
    sched[idx++] = 2; }

  const int ldaS = SSq;           // stack_h row stride
  const int ldaB = NTRq * Hq;     // bufH batch-row stride

  int p = 2, k = 0, r = 0;
  for (int t = 0; t < NTRq; ++t) {
    // tracking gates: 4 segments, M=B, N=4T
    {
      int blocks = (Bq / 128) * (GN / 64);   // 8*16 = 128
      k_gemm4<<<blocks, 256, 0, stream>>>(
          bufH + k * Hq,           ldaB, Wih_bf + 0 * Hq, 3 * Hq, Hq,
          stack_h + (p - 1) * Hq,  ldaS, Wih_bf + 1 * Hq, 3 * Hq, Hq,
          stack_h + (p - 2) * Hq,  ldaS, Wih_bf + 2 * Hq, 3 * Hq, Hq,
          th,                      Tq,   Whh_bf,          Tq,     Tq,
          Bq, GN, b_ih, b_hh, gates);
    }
    k_lstm<<<(Bq * Tq + 255) / 256, 256, 0, stream>>>(gates, tc, th);

    if (sched[t] == 1) {  // SHIFT
      k_shift<<<(Bq * Hq + 255) / 256, 256, 0, stream>>>(bufH, k, p, stack_h, stack_c);
      ++p; ++k;
    } else {              // REDUCE
      int blocks = (Bq / 128) * (RN / 64);   // 8*20 = 160
      k_gemm4<<<blocks, 256, 0, stream>>>(
          stack_h + (p - 2) * Hq, ldaS, Lw_bf, Hq, Hq,
          stack_h + (p - 1) * Hq, ldaS, Rw_bf, Hq, Hq,
          th,                     Tq,   Tw_bf, Tq, Tq,
          (const bf16_t*)nullptr, 0, (const bf16_t*)nullptr, 0, 0,
          Bq, RN, left_b, (const float*)nullptr, lstm_in);
      k_reduce<<<(Bq * Hq + 255) / 256, 256, 0, stream>>>(lstm_in, p, stack_h, stack_c);
      --p;
      k_out1<<<(Bq + 255) / 256, 256, 0, stream>>>(stack_h, p + 1, W_out_w, W_out_b, out1, r);
      ++r;
    }
  }
}
